// GearNetFromCoordinates_68856915690070
// MI455X (gfx1250) — matrix operand order repaired and checked by reading, not compiled, not run on hardware
//
#include <hip/hip_runtime.h>
#include <hip/hip_bf16.h>
#include <stdint.h>

// ---------------- problem constants ----------------
#define BB      8
#define LL      2048
#define NNODE   (BB * LL)        // 16384
#define HID     512
#define KNN     10
#define KCAT    1536             // [agg_seq | agg_knn | self] each 512
#define BN_EPS  1e-5f

typedef __attribute__((ext_vector_type(16))) __bf16 v16bf;
typedef __attribute__((ext_vector_type(8)))  __bf16 v8bf;
typedef __attribute__((ext_vector_type(8)))  float  v8f;

__device__ __forceinline__ __bf16 f2bf(float f) { return (__bf16)f; }

// CDNA5 async copy: global memory -> LDS, 16B per lane, tracked by ASYNCcnt.
// dst operand is the LDS byte address (low 32 bits of a generic LDS pointer).
__device__ __forceinline__ void async_cp16(void* lds_dst, const void* gsrc) {
    unsigned lds_off = (unsigned)(uintptr_t)lds_dst;
    asm volatile("global_load_async_to_lds_b128 %0, %1, off"
                 :: "v"(lds_off), "v"(gsrc) : "memory");
}

// ---------------- utility: zero fp32 buffer ----------------
__global__ void zero_f32_kernel(float* __restrict__ p, int n) {
    int i = blockIdx.x * blockDim.x + threadIdx.x;
    if (i < n) p[i] = 0.0f;
}

// ---------------- KNN: per node, 10 nearest in its batch row ----------------
__global__ void knn_kernel(const float* __restrict__ ca, int* __restrict__ knn_out) {
    int n = blockIdx.x * blockDim.x + threadIdx.x;
    if (n >= NNODE) return;
    int b = n / LL, i = n % LL;
    const float* base = ca + (size_t)b * LL * 3;
    float xi = base[i * 3 + 0], yi = base[i * 3 + 1], zi = base[i * 3 + 2];

    float bd[KNN];
    int   bi[KNN];
#pragma unroll
    for (int k = 0; k < KNN; ++k) { bd[k] = 3.4e38f; bi[k] = 0; }

    for (int j = 0; j < LL; ++j) {
        if (j == i) continue;
        float dx = base[j * 3 + 0] - xi;
        float dy = base[j * 3 + 1] - yi;
        float dz = base[j * 3 + 2] - zi;
        float d2 = dx * dx + dy * dy + dz * dz;
        if (d2 < bd[KNN - 1]) {
            int p = KNN - 1;
            while (p > 0 && d2 < bd[p - 1]) { bd[p] = bd[p - 1]; bi[p] = bi[p - 1]; --p; }
            bd[p] = d2; bi[p] = j;
        }
    }
#pragma unroll
    for (int k = 0; k < KNN; ++k) knn_out[n * KNN + k] = b * LL + bi[k];
}

// ---------------- coords: sequential (gather) aggregation ----------------
__global__ void seqagg_coords_kernel(const float* __restrict__ ca, float* __restrict__ out) {
    int n = blockIdx.x * blockDim.x + threadIdx.x;
    if (n >= NNODE) return;
    int i = n % LL;
    float s0 = 0.f, s1 = 0.f, s2 = 0.f;
#pragma unroll
    for (int off = 1; off <= 3; ++off) {
        if (i - off >= 0) {
            s0 += ca[(n - off) * 3 + 0]; s1 += ca[(n - off) * 3 + 1]; s2 += ca[(n - off) * 3 + 2];
        }
        if (i + off < LL) {
            s0 += ca[(n + off) * 3 + 0]; s1 += ca[(n + off) * 3 + 1]; s2 += ca[(n + off) * 3 + 2];
        }
    }
    out[n * 3 + 0] = s0; out[n * 3 + 1] = s1; out[n * 3 + 2] = s2;
}

// ---------------- coords: spatial scatter (src -> each of its KNN dsts) ----------------
__global__ void scatter_coords_kernel(const float* __restrict__ ca, const int* __restrict__ knn,
                                      float* __restrict__ agg) {
    int e = blockIdx.x * blockDim.x + threadIdx.x;
    if (e >= NNODE * KNN) return;
    int src = e / KNN;
    int dst = knn[e];
    atomicAdd(&agg[dst * 3 + 0], ca[src * 3 + 0]);
    atomicAdd(&agg[dst * 3 + 1], ca[src * 3 + 1]);
    atomicAdd(&agg[dst * 3 + 2], ca[src * 3 + 2]);
}

// ---------------- layer 1: din = 3 (plain VALU, memory-trivial) ----------------
__global__ void layer1_kernel(const float* __restrict__ ca,
                              const float* __restrict__ aggseq,
                              const float* __restrict__ aggsp,
                              const float* __restrict__ W1,   // (512, 21)
                              const float* __restrict__ b1,
                              const float* __restrict__ Wsl1, // (512, 3)
                              const float* __restrict__ bsl1,
                              const float* __restrict__ gamma,
                              const float* __restrict__ beta,
                              const float* __restrict__ mean,
                              const float* __restrict__ var,
                              float* __restrict__ hout) {
    int idx = blockIdx.x * blockDim.x + threadIdx.x;
    if (idx >= NNODE * HID) return;
    int n = idx >> 9;
    int c = idx & (HID - 1);
    float pre = b1[c] + bsl1[c];
#pragma unroll
    for (int d = 0; d < 3; ++d) {
        pre += aggseq[n * 3 + d] * W1[c * 21 + d];       // rel 0 block
        pre += aggsp[n * 3 + d]  * W1[c * 21 + 6 + d];   // rel 2 block
        pre += ca[n * 3 + d]     * Wsl1[c * 3 + d];      // self loop
    }
    float s = gamma[c] * rsqrtf(var[c] + BN_EPS);
    pre = (pre - mean[c]) * s + beta[c];
    hout[(size_t)n * HID + c] = fmaxf(pre, 0.0f);
}

// ---------------- hidden: spatial scatter (fp32 atomics) ----------------
__global__ void scatter_h_kernel(const float* __restrict__ h, const int* __restrict__ knn,
                                 float* __restrict__ agg) {
    int idx = blockIdx.x * blockDim.x + threadIdx.x;       // (edge, 128 chan-groups of 4)
    if (idx >= NNODE * KNN * (HID / 4)) return;
    int e  = idx >> 7;
    int cg = (idx & 127) * 4;
    int src = e / KNN;
    int dst = knn[e];
    const float4 v = *(const float4*)&h[(size_t)src * HID + cg];
    float* d = &agg[(size_t)dst * HID + cg];
    atomicAdd(d + 0, v.x);
    atomicAdd(d + 1, v.y);
    atomicAdd(d + 2, v.z);
    atomicAdd(d + 3, v.w);
}

// ---------------- pack bf16 A-panel: [seq-gather | knn-agg | self] ----------------
__global__ void pack_kernel(const float* __restrict__ h, const float* __restrict__ agg2,
                            __bf16* __restrict__ A) {
    int idx = blockIdx.x * blockDim.x + threadIdx.x;
    if (idx >= NNODE * HID) return;
    int n = idx >> 9;
    int c = idx & (HID - 1);
    int i = n % LL;
    float s = 0.f;
#pragma unroll
    for (int off = 1; off <= 3; ++off) {
        if (i - off >= 0) s += h[(size_t)(n - off) * HID + c];
        if (i + off < LL) s += h[(size_t)(n + off) * HID + c];
    }
    __bf16* row = A + (size_t)n * KCAT;
    row[c]            = f2bf(s);
    row[HID + c]      = f2bf(agg2[(size_t)n * HID + c]);
    row[2 * HID + c]  = f2bf(h[(size_t)n * HID + c]);
}

// ---------------- pack bf16 concatenated weights (512 x 1536, N-major) ----------------
__global__ void packW_kernel(const float* __restrict__ W,   // (512, 3584)
                             const float* __restrict__ Wsl, // (512, 512)
                             __bf16* __restrict__ Wc) {
    int idx = blockIdx.x * blockDim.x + threadIdx.x;
    if (idx >= HID * KCAT) return;
    int r = idx / KCAT;
    int k = idx - r * KCAT;
    float v;
    if (k < HID)           v = W[(size_t)r * (7 * HID) + k];                   // rel 0
    else if (k < 2 * HID)  v = W[(size_t)r * (7 * HID) + 2 * HID + (k - HID)]; // rel 2
    else                   v = Wsl[(size_t)r * HID + (k - 2 * HID)];           // self
    Wc[(size_t)r * KCAT + k] = f2bf(v);
}

// ---------------- per-column BN affine precompute ----------------
__global__ void colparam_kernel(const float* __restrict__ b, const float* __restrict__ bsl,
                                const float* __restrict__ gamma, const float* __restrict__ beta,
                                const float* __restrict__ mean, const float* __restrict__ var,
                                float* __restrict__ colA, float* __restrict__ colC) {
    int c = blockIdx.x * blockDim.x + threadIdx.x;
    if (c >= HID) return;
    float a = gamma[c] * rsqrtf(var[c] + BN_EPS);
    colA[c] = a;
    colC[c] = (b[c] + bsl[c] - mean[c]) * a + beta[c];
}

// ---------------- fused WMMA GEMM + BN + ReLU + residual ----------------
// out[16384,512] = relu( (A[16384,1536] @ Wc.T) * colA + colC ) + resid
// block = 256 threads (8 waves); block tile 128(M) x 64(N); wave tile 32x32 = 2x2 wmma.
// Double-buffered LDS, tiles staged with GLOBAL_LOAD_ASYNC_TO_LDS_B128 (ASYNCcnt).
// Last K-tile is peeled so the steady-state loop is branch-free:
//   stage(next) -> s_wait_asynccnt 3 -> barrier -> 8x ds_load_b128 + 4x wmma -> barrier
#define LDSTRIDE 40  // 32 + 8 bf16 pad

__global__ void __launch_bounds__(256)
gemm_wmma_kernel(const __bf16* __restrict__ A, const __bf16* __restrict__ Wc,
                 const float* __restrict__ colA, const float* __restrict__ colC,
                 const float* __restrict__ resid, float* __restrict__ out) {
    __shared__ __align__(16) __bf16 As[2][128 * LDSTRIDE];
    __shared__ __align__(16) __bf16 Bs[2][64 * LDSTRIDE];

    const int t    = threadIdx.x;
    const int lane = t & 31;
    const int wv   = t >> 5;       // 0..7
    const int wm   = wv >> 1;      // 0..3 -> M sub-tile
    const int wn   = wv & 1;       // 0..1 -> N sub-tile
    const int g    = lane >> 4;    // lane half
    const int lm   = lane & 15;

    const int mt = blockIdx.x * 128;
    const int nt = blockIdx.y * 64;

    // Per-thread staging assignments (3 async b128 loads per thread per tile):
    //   A tile: 128 rows x 32 k -> 512 16B chunks, chunks t and t+256
    //   B tile:  64 rows x 32 k -> 256 16B chunks, chunk t
    const int a0row = t >> 2,         a0kc = t & 3;
    const int a1row = (t + 256) >> 2, a1kc = (t + 256) & 3;
    const int brow  = t >> 2,         bkc  = t & 3;

    auto stage = [&](int buf, int kt) {
        async_cp16(&As[buf][a0row * LDSTRIDE + a0kc * 8],
                   A + (size_t)(mt + a0row) * KCAT + kt + a0kc * 8);
        async_cp16(&As[buf][a1row * LDSTRIDE + a1kc * 8],
                   A + (size_t)(mt + a1row) * KCAT + kt + a1kc * 8);
        async_cp16(&Bs[buf][brow * LDSTRIDE + bkc * 8],
                   Wc + (size_t)(nt + brow) * KCAT + kt + bkc * 8);
    };

    v8f acc[2][2] = {};

    // math on buffer `cur` (fragments per ISA 7.12.2 layouts)
    auto compute = [&](int cur) {
        v16bf afr[2], bfr[2];
#pragma unroll
        for (int i = 0; i < 2; ++i) {
            const __bf16* base = &As[cur][(wm * 32 + i * 16 + lm) * LDSTRIDE];
            v8bf lo = *(const v8bf*)(base + g * 8);        // K = g*8 + 0..7
            v8bf hi = *(const v8bf*)(base + 16 + g * 8);   // K = 16 + g*8 + 0..7
            afr[i] = __builtin_shufflevector(lo, hi, 0, 1, 2, 3, 4, 5, 6, 7,
                                                     8, 9, 10, 11, 12, 13, 14, 15);
        }
#pragma unroll
        for (int j = 0; j < 2; ++j) {
            const __bf16* base = &Bs[cur][(wn * 32 + j * 16 + lm) * LDSTRIDE];
            v8bf lo = *(const v8bf*)(base + g * 8);        // K = g*8 + 0..7
            v8bf hi = *(const v8bf*)(base + 16 + g * 8);   // K = 16 + g*8 + 0..7
            bfr[j] = __builtin_shufflevector(lo, hi, 0, 1, 2, 3, 4, 5, 6, 7,
                                                     8, 9, 10, 11, 12, 13, 14, 15);
        }
#pragma unroll
        for (int i = 0; i < 2; ++i)
#pragma unroll
            for (int j = 0; j < 2; ++j)
                acc[i][j] = __builtin_amdgcn_wmma_f32_16x16x32_bf16(
                    false, afr[i], false, bfr[j], (short)0, acc[i][j], false, false);
    };

    stage(0, 0);                       // prologue: fill buffer 0
    int cur = 0;

    // steady state: always stage next tile, always wait for the older batch of 3.
    for (int kt = 0; kt < KCAT - 32; kt += 32) {
        stage(cur ^ 1, kt + 32);
        // async loads complete in order: <=3 outstanding means buffer `cur` is ready.
        asm volatile("s_wait_asynccnt 0x3" ::: "memory");
        __syncthreads();
        compute(cur);
        __syncthreads();               // all reads of `cur` done before it is restaged
        cur ^= 1;
    }

    // peeled final tile
    asm volatile("s_wait_asynccnt 0x0" ::: "memory");
    __syncthreads();
    compute(cur);

    // ---- epilogue: BN affine + ReLU + residual ----
#pragma unroll
    for (int i = 0; i < 2; ++i) {
#pragma unroll
        for (int j = 0; j < 2; ++j) {
            const int col = nt + wn * 32 + j * 16 + lm;
            const float a = colA[col];
            const float c0 = colC[col];
#pragma unroll
            for (int r = 0; r < 8; ++r) {
                const int row = mt + wm * 32 + i * 16 + g * 8 + r;
                float v = acc[i][j][r] * a + c0;
                v = fmaxf(v, 0.0f) + resid[(size_t)row * HID + col];
                out[(size_t)row * HID + col] = v;
            }
        }
    }
}

// ---------------- host-side orchestration ----------------
extern "C" void kernel_launch(void* const* d_in, const int* in_sizes, int n_in,
                              void* d_out, int out_size, void* d_ws, size_t ws_size,
                              hipStream_t stream) {
    (void)in_sizes; (void)n_in; (void)out_size; (void)ws_size;

    const float* ca    = (const float*)d_in[1];
    const float* W1    = (const float*)d_in[3];
    const float* b1    = (const float*)d_in[4];
    const float* Wsl1  = (const float*)d_in[5];
    const float* bsl1  = (const float*)d_in[6];
    const float* Ws    = (const float*)d_in[7];
    const float* bs    = (const float*)d_in[8];
    const float* Wsls  = (const float*)d_in[9];
    const float* bsls  = (const float*)d_in[10];
    const float* gam   = (const float*)d_in[11];
    const float* bet   = (const float*)d_in[12];
    const float* mea   = (const float*)d_in[13];
    const float* var   = (const float*)d_in[14];

    char* ws = (char*)d_ws;
    size_t off = 0;
    auto carve = [&](size_t bytes) { char* p = ws + off; off += (bytes + 255) & ~(size_t)255; return p; };

    int*    knn     = (int*)   carve((size_t)NNODE * KNN * 4);
    float*  caggseq = (float*) carve((size_t)NNODE * 3 * 4);
    float*  caggsp  = (float*) carve((size_t)NNODE * 3 * 4);
    float*  agg2f   = (float*) carve((size_t)NNODE * HID * 4);
    __bf16* Apanel  = (__bf16*)carve((size_t)NNODE * KCAT * 2);
    __bf16* Wcat    = (__bf16*)carve((size_t)HID * KCAT * 2);
    float*  colAv   = (float*) carve((size_t)HID * 4);
    float*  colCv   = (float*) carve((size_t)HID * 4);
    float*  hA      = (float*) carve((size_t)NNODE * HID * 4);
    float*  hB      = (float*) carve((size_t)NNODE * HID * 4);

    const int TB = 256;

    // graph build
    knn_kernel<<<(NNODE + TB - 1) / TB, TB, 0, stream>>>(ca, knn);
    zero_f32_kernel<<<(NNODE * 3 + TB - 1) / TB, TB, 0, stream>>>(caggsp, NNODE * 3);
    seqagg_coords_kernel<<<(NNODE + TB - 1) / TB, TB, 0, stream>>>(ca, caggseq);
    scatter_coords_kernel<<<(NNODE * KNN + TB - 1) / TB, TB, 0, stream>>>(ca, knn, caggsp);

    // layer 1 (coords -> 512), BN row 0
    layer1_kernel<<<(NNODE * HID) / TB, TB, 0, stream>>>(
        ca, caggseq, caggsp, W1, b1, Wsl1, bsl1,
        gam + 0, bet + 0, mea + 0, var + 0, hA);

    // 3 residual layers
    for (int i = 0; i < 3; ++i) {
        float* hcur = (i == 1) ? hB : hA;
        float* hnxt = (i == 2) ? (float*)d_out : ((i == 0) ? hB : hA);

        zero_f32_kernel<<<(NNODE * HID) / TB, TB, 0, stream>>>(agg2f, NNODE * HID);
        scatter_h_kernel<<<(NNODE * KNN * (HID / 4)) / TB, TB, 0, stream>>>(hcur, knn, agg2f);
        pack_kernel<<<(NNODE * HID) / TB, TB, 0, stream>>>(hcur, agg2f, Apanel);
        packW_kernel<<<(HID * KCAT) / TB, TB, 0, stream>>>(
            Ws + (size_t)i * HID * (7 * HID), Wsls + (size_t)i * HID * HID, Wcat);
        colparam_kernel<<<(HID + TB - 1) / TB, TB, 0, stream>>>(
            bs + (size_t)i * HID, bsls + (size_t)i * HID,
            gam + (size_t)(i + 1) * HID, bet + (size_t)(i + 1) * HID,
            mea + (size_t)(i + 1) * HID, var + (size_t)(i + 1) * HID,
            colAv, colCv);

        dim3 grid(NNODE / 128, HID / 64);
        gemm_wmma_kernel<<<grid, TB, 0, stream>>>(Apanel, Wcat, colAv, colCv, hcur, hnxt);
    }
}
